// net_68985764708965
// MI455X (gfx1250) — compile-verified, matrix operand order checked by reading, not run on hardware
//
#include <hip/hip_runtime.h>
#include <hip/hip_bf16.h>

typedef _Float16 h16 __attribute__((ext_vector_type(16)));
typedef _Float16 h8  __attribute__((ext_vector_type(8)));
typedef _Float16 h4  __attribute__((ext_vector_type(4)));
typedef float    f8  __attribute__((ext_vector_type(8)));
typedef float    f4v __attribute__((ext_vector_type(4)));

#define T_STEPS 2048
#define BATCH   128
#define IN_DIM  64
#define HID     128
#define G4      512          // 4*H
#define ROWS_PER_WG 16
#define THREADS 256

// ---- LDS layout (bytes) ----
#define SM_WHH   0
#define SM_WIH   (SM_WHH + G4*HID*2)          // 131072
#define SM_H     (SM_WIH + G4*IN_DIM*2)       // + 65536
#define SM_X     (SM_H   + 2*ROWS_PER_WG*HID*2)     // + 8192  (double buffered)
#define SM_HNEW  (SM_X   + 2*ROWS_PER_WG*IN_DIM*2)  // + 4096  (double buffered)
#define SM_HID   (SM_HNEW + ROWS_PER_WG*384*4)      // + 24576
#define SM_TOTAL (SM_HID  + ROWS_PER_WG*HID*4)      // + 8192  => 241664 B < 320 KB

__device__ __forceinline__ h16 ld_op(const _Float16* p0) {
  // assemble a v16h WMMA operand from two contiguous 8-half chunks
  union { h16 v; h8 h[2]; } u;
  u.h[0] = *(const h8*)(p0);
  u.h[1] = *(const h8*)(p0 + 16);
  return u.v;
}

// CDNA5 native v_tanh_f32 (probe-confirmed to lower on gfx1250)
__device__ __forceinline__ float tanh_fast(float x) {
#if __has_builtin(__builtin_amdgcn_tanhf)
  return __builtin_amdgcn_tanhf(x);                       // 1 TRANS op
#else
  return 1.0f - 2.0f * __builtin_amdgcn_rcpf(__expf(2.0f * x) + 1.0f);
#endif
}
__device__ __forceinline__ float sigf(float x) {
#if __has_builtin(__builtin_amdgcn_tanhf)
  return fmaf(0.5f, __builtin_amdgcn_tanhf(0.5f * x), 0.5f);
#else
  return __builtin_amdgcn_rcpf(1.0f + __expf(-x));
#endif
}

__global__ __launch_bounds__(THREADS)
void lstm_pool_mlp(const float* __restrict__ batch, const int* __restrict__ sizes,
                   const float* __restrict__ W_ih, const float* __restrict__ W_hh,
                   const float* __restrict__ b_ih, const float* __restrict__ b_hh,
                   const float* __restrict__ W1,   const float* __restrict__ b1,
                   const float* __restrict__ W2,   const float* __restrict__ b2,
                   float* __restrict__ out)
{
  extern __shared__ __align__(128) char smem[];
  _Float16* sWhh = (_Float16*)(smem + SM_WHH);   // [512][128] row-major (gate-col major)
  _Float16* sWih = (_Float16*)(smem + SM_WIH);   // [512][64]
  _Float16* sH   = (_Float16*)(smem + SM_H);     // 2 x [16][128]
  _Float16* sX   = (_Float16*)(smem + SM_X);     // 2 x [16][64]
  float*    sHnew= (float*)   (smem + SM_HNEW);  // [16][384]
  float*    sHid = (float*)   (smem + SM_HID);   // [16][128]

  const int tid  = threadIdx.x;
  const int lane = tid & 31;
  const int wid  = tid >> 5;            // 0..7 : hidden-column tile
  const int rb0  = blockIdx.x * ROWS_PER_WG;

  // ---- stage weights fp32 -> fp16 into LDS; zero h buffer 0; stage x_0 ----
  for (int i = tid; i < G4*HID;          i += THREADS) sWhh[i] = (_Float16)W_hh[i];
  for (int i = tid; i < G4*IN_DIM;       i += THREADS) sWih[i] = (_Float16)W_ih[i];
  for (int i = tid; i < ROWS_PER_WG*HID; i += THREADS) sH[i] = (_Float16)0.0f;
  {
    const float* x0 = batch + (size_t)rb0 * IN_DIM;
    const f4v xv = *(const f4v*)(x0 + tid * 4);
    h4 xh;
#pragma unroll
    for (int k = 0; k < 4; ++k) xh[k] = (_Float16)xv[k];
    *(h4*)(sX + tid * 4) = xh;        // buffer 0 holds x_0
  }
  __syncthreads();

  const int nloc = lane & 15;           // N index within tile (16-bit A/B/C lane col)
  const int hi   = (lane >> 4) & 1;     // which half of the wave
  const int kbA  = hi ? 8 : 0;          // K sub-offset for 16-bit operand layout
  const int nH   = wid * 16;            // hidden-column base of this wave's tile

  // ---- pin fp16 weight B-operands in VGPRs for the whole scan ----
  h16 Bh[4][4];                         // [gate][K-window of 32] over W_hh (K=128)
  h16 Bx[4][2];                         // [gate][K-window of 32] over W_ih (K=64)
#pragma unroll
  for (int g = 0; g < 4; ++g) {
    const int ncol = g * HID + nH + nloc;       // row of W (= column of B)
#pragma unroll
    for (int kw = 0; kw < 4; ++kw)
      Bh[g][kw] = ld_op(&sWhh[ncol * HID + kw * 32 + kbA]);
#pragma unroll
    for (int kw = 0; kw < 2; ++kw)
      Bx[g][kw] = ld_op(&sWih[ncol * IN_DIM + kw * 32 + kbA]);
  }

  // folded gate bias (b_ih + b_hh) splatted into f8 seeds for the x-projection
  f8 bias_v[4];
#pragma unroll
  for (int g = 0; g < 4; ++g) {
    const int col = g * HID + nH + nloc;
    const float b = b_ih[col] + b_hh[col];
#pragma unroll
    for (int r = 0; r < 8; ++r) bias_v[g][r] = b;
  }

  // per-element (batch-row) sequence lengths minus one; C-layout row = r + 8*hi
  int szm1[8];
#pragma unroll
  for (int r = 0; r < 8; ++r) szm1[r] = sizes[rb0 + r + hi * 8] - 1;

  float cst[8], hmax[8], hsum[8], hlast[8];
#pragma unroll
  for (int r = 0; r < 8; ++r) {
    cst[r] = 0.0f; hsum[r] = 0.0f; hlast[r] = 0.0f; hmax[r] = -__builtin_inff();
  }

  // ---- prologue of the software pipeline: xacc_0 = bias + x_0 @ W_ih^T ----
  f8 xacc[4];
  {
    const _Float16* rowX = &sX[nloc * IN_DIM];          // buffer 0
    const h16 Ax0 = ld_op(rowX + kbA);
    const h16 Ax1 = ld_op(rowX + 32 + kbA);
#pragma unroll
    for (int g = 0; g < 4; ++g) {
      f8 a = __builtin_amdgcn_wmma_f32_16x16x32_f16(false, Ax0, false, Bx[g][0],
                                                    (short)0, bias_v[g], false, false);
      xacc[g] = __builtin_amdgcn_wmma_f32_16x16x32_f16(false, Ax1, false, Bx[g][1],
                                                       (short)0, a, false, false);
    }
  }

  // ================= sequential scan over T =================
#pragma unroll 1
  for (int t = 0; t < T_STEPS; ++t) {
    const int buf  = t & 1;
    const int nbuf = buf ^ 1;
    _Float16* sHin  = sH + buf  * (ROWS_PER_WG * HID);
    _Float16* sHout = sH + nbuf * (ROWS_PER_WG * HID);
    _Float16* sXn   = sX + nbuf * (ROWS_PER_WG * IN_DIM);

    // stage x_{t+1} (pre-barrier); consumed post-barrier this iteration
    if (t + 1 < T_STEPS) {
      const float* xt = batch + ((size_t)(t + 1) * BATCH + rb0) * IN_DIM;
      const f4v xv = *(const f4v*)(xt + tid * 4);
      h4 xh;
#pragma unroll
      for (int k = 0; k < 4; ++k) xh[k] = (_Float16)xv[k];
      *(h4*)(sXn + tid * 4) = xh;
      if (t + 2 < T_STEPS) {
        const float* xn = batch + ((size_t)(t + 2) * BATCH + rb0) * IN_DIM;
        __builtin_prefetch(xn + tid * 4, 0, 1);   // global_prefetch
      }
    }
    __syncthreads();   // h_t (written last step) + x_{t+1} now visible

    // recurrent critical path: only 4 chained WMMAs, seeded from xacc_t
    h16 Ah[4];
    {
      const _Float16* rowH = &sHin[nloc * HID];
#pragma unroll
      for (int kw = 0; kw < 4; ++kw) Ah[kw] = ld_op(rowH + kw * 32 + kbA);
    }
    f8 acc[4];
#pragma unroll
    for (int g = 0; g < 4; ++g) {
      f8 a = __builtin_amdgcn_wmma_f32_16x16x32_f16(false, Ah[0], false, Bh[g][0],
                                                    (short)0, xacc[g], false, false);
#pragma unroll
      for (int kw = 1; kw < 4; ++kw)
        a = __builtin_amdgcn_wmma_f32_16x16x32_f16(false, Ah[kw], false, Bh[g][kw],
                                                   (short)0, a, false, false);
      acc[g] = a;
    }

    // off-path: xacc_{t+1} = bias + x_{t+1} @ W_ih^T (independent of h-chain)
    if (t + 1 < T_STEPS) {
      const _Float16* rowX = &sXn[nloc * IN_DIM];
      const h16 Ax0 = ld_op(rowX + kbA);
      const h16 Ax1 = ld_op(rowX + 32 + kbA);
#pragma unroll
      for (int g = 0; g < 4; ++g) {
        f8 a = __builtin_amdgcn_wmma_f32_16x16x32_f16(false, Ax0, false, Bx[g][0],
                                                      (short)0, bias_v[g], false, false);
        xacc[g] = __builtin_amdgcn_wmma_f32_16x16x32_f16(false, Ax1, false, Bx[g][1],
                                                         (short)0, a, false, false);
      }
    }

    // LSTM cell + masked pooling, write h back for next step
#pragma unroll
    for (int r = 0; r < 8; ++r) {
      const float gi = sigf(acc[0][r]);
      const float gf = sigf(acc[1][r]);
      const float gg = tanh_fast(acc[2][r]);
      const float go = sigf(acc[3][r]);
      const float c  = gf * cst[r] + gi * gg;
      cst[r] = c;
      const float h = go * tanh_fast(c);

      const bool m = (t < szm1[r]);
      hmax[r]  = m ? fmaxf(hmax[r], h) : hmax[r];
      hsum[r] += m ? h : 0.0f;
      hlast[r] = (t == szm1[r]) ? h : hlast[r];

      sHout[(r + hi * 8) * HID + nH + nloc] = (_Float16)h;
    }
  }

  // ---- pooled features: [h_last | h_max | h_mean] -> sHnew[16][384] ----
#pragma unroll
  for (int r = 0; r < 8; ++r) {
    const int row = r + hi * 8;
    const int col = nH + nloc;
    sHnew[row * 384 + col]       = hlast[r];
    sHnew[row * 384 + 128 + col] = hmax[r];
    sHnew[row * 384 + 256 + col] = hsum[r] / (float)szm1[r];
  }
  __syncthreads();

  // ---- MLP layer 1: hid = leaky_relu(h_new @ W1^T + b1), fp32 ----
  for (int o = tid; o < ROWS_PER_WG * HID; o += THREADS) {
    const int row = o >> 7;
    const int j   = o & 127;
    const float* w  = &W1[j * 384];
    const float* hn = &sHnew[row * 384];
    float a = b1[j];
#pragma unroll 4
    for (int k = 0; k < 384; ++k) a = fmaf(hn[k], w[k], a);
    sHid[row * HID + j] = a > 0.0f ? a : 0.01f * a;
  }
  __syncthreads();

  // ---- MLP layer 2: out = hid @ W2^T + b2 ----
  if (tid < ROWS_PER_WG * 2) {
    const int row = tid >> 1;
    const int j   = tid & 1;
    const float* w  = &W2[j * HID];
    const float* hd = &sHid[row * HID];
    float a = b2[j];
#pragma unroll 4
    for (int k = 0; k < HID; ++k) a = fmaf(hd[k], w[k], a);
    out[(rb0 + row) * 2 + j] = a;
  }
}

extern "C" void kernel_launch(void* const* d_in, const int* in_sizes, int n_in,
                              void* d_out, int out_size, void* d_ws, size_t ws_size,
                              hipStream_t stream) {
  const float* batch = (const float*)d_in[0];
  const int*   sizes = (const int*)  d_in[1];
  const float* W_ih  = (const float*)d_in[2];
  const float* W_hh  = (const float*)d_in[3];
  const float* b_ih  = (const float*)d_in[4];
  const float* b_hh  = (const float*)d_in[5];
  const float* W1    = (const float*)d_in[6];
  const float* b1    = (const float*)d_in[7];
  const float* W2    = (const float*)d_in[8];
  const float* b2    = (const float*)d_in[9];
  float* out = (float*)d_out;

  dim3 grid(BATCH / ROWS_PER_WG);   // 8 independent workgroups, one per WGP
  dim3 block(THREADS);              // 8 waves of 32
  hipLaunchKernelGGL(lstm_pool_mlp, grid, block, SM_TOTAL, stream,
                     batch, sizes, W_ih, W_hh, b_ih, b_hh, W1, b1, W2, b2, out);
}
